// Transformer_3092376453758
// MI455X (gfx1250) — matrix operand order repaired and checked by reading, not compiled, not run on hardware
//
#include <hip/hip_runtime.h>

// ---------------------------------------------------------------------------
// Types for CDNA5 WMMA (wave32): v_wmma_f32_16x16x32_bf16
// ---------------------------------------------------------------------------
typedef __attribute__((ext_vector_type(16))) __bf16 v16bf;
typedef __attribute__((ext_vector_type(8)))  float  v8f;
typedef __attribute__((ext_vector_type(4)))  int    v4i;

union VecBF {
    v16bf v;
    unsigned short s[16];
    uint4 q[2];
};

__device__ __forceinline__ v8f wmma_bf16(const VecBF& a, const VecBF& b, v8f c) {
    // 8 args: (neg_a, A, neg_b, B, c_mod, C, reuse_a, reuse_b)
    return __builtin_amdgcn_wmma_f32_16x16x32_bf16(false, a.v, false, b.v,
                                                   (short)0, c, false, false);
}

__device__ __forceinline__ unsigned short f2bf(float x) {
    union { float f; unsigned u; } v;
    v.f = x;
    unsigned u = v.u;
    unsigned r = u + 0x7fffu + ((u >> 16) & 1u);  // round-to-nearest-even
    return (unsigned short)(r >> 16);
}

// ---------------------------------------------------------------------------
// Async global->LDS staging (CDNA5 GLOBAL_LOAD_ASYNC_TO_LDS_B128, ASYNCcnt).
// Signature (from the toolchain's diagnostic): (v4i AS1*, v4i AS3*, Ii, Ii).
// Graceful fallback to plain uint4 copies if the builtin is absent.
// ---------------------------------------------------------------------------
#if __has_builtin(__builtin_amdgcn_global_load_async_to_lds_b128)
#define ATH_ASYNC_CP 1
typedef __attribute__((address_space(1))) v4i* gptr4_t;
typedef __attribute__((address_space(3))) v4i* lptr4_t;
#endif

__device__ __forceinline__ void cp16_g2l(void* lds, const void* g) {
#if defined(ATH_ASYNC_CP)
    __builtin_amdgcn_global_load_async_to_lds_b128((gptr4_t)g, (lptr4_t)lds, 0, 0);
#else
    *(uint4*)lds = *(const uint4*)g;
#endif
}

__device__ __forceinline__ void cp_wait() {
#if defined(ATH_ASYNC_CP)
#if __has_builtin(__builtin_amdgcn_s_wait_asynccnt)
    __builtin_amdgcn_s_wait_asynccnt(0);
#else
    asm volatile("s_wait_asynccnt 0x0" ::: "memory");
#endif
#endif
}

// ---------------------------------------------------------------------------
// DPP16 butterfly reductions across a 16-lane group (pure VALU, no LDS).
// quad_perm(1,0,3,2)=0xB1, quad_perm(2,3,0,1)=0x4E,
// row_half_mirror=0x141, row_mirror=0x140.
// ---------------------------------------------------------------------------
template <int CTRL>
__device__ __forceinline__ float dpp_movf(float x) {
    int i = __builtin_bit_cast(int, x);
    int r = __builtin_amdgcn_update_dpp(i, i, CTRL, 0xf, 0xf, true);
    return __builtin_bit_cast(float, r);
}

__device__ __forceinline__ float red16_max(float x) {
    x = fmaxf(x, dpp_movf<0xB1>(x));
    x = fmaxf(x, dpp_movf<0x4E>(x));
    x = fmaxf(x, dpp_movf<0x141>(x));
    x = fmaxf(x, dpp_movf<0x140>(x));
    return x;
}

__device__ __forceinline__ float red16_sum(float x) {
    x += dpp_movf<0xB1>(x);
    x += dpp_movf<0x4E>(x);
    x += dpp_movf<0x141>(x);
    x += dpp_movf<0x140>(x);
    return x;
}

// ---------------------------------------------------------------------------
// f32 -> bf16 cast (weights & initial x)
// ---------------------------------------------------------------------------
__global__ __launch_bounds__(256) void cast_bf16_kernel(const float* __restrict__ src,
                                                        unsigned short* __restrict__ dst,
                                                        int n) {
    int i = blockIdx.x * 256 + threadIdx.x;
    if (i < n) dst[i] = f2bf(src[i]);
}

// ---------------------------------------------------------------------------
// Tiled bf16 GEMM with WMMA:  C[M,N] = act(A[M,K] * W[K,N] + bias[N])
// Block tile 128x128, 8 waves (4m x 2n), each wave 32x64:
//   2 A-fragments x 4 B-fragments -> 8 WMMAs per 32-wide K step.
// ---------------------------------------------------------------------------
template <int RELU, int WF, int WB>
__global__ __launch_bounds__(256) void gemm_bf16_kernel(const unsigned short* __restrict__ A,
                                                        const unsigned short* __restrict__ W,
                                                        const float* __restrict__ bias,
                                                        unsigned short* __restrict__ outB,
                                                        float* __restrict__ outF,
                                                        int M, int N, int K) {
    __shared__ __align__(16) unsigned short As[128][32];  // [m][k]
    __shared__ __align__(16) unsigned short Bt[128][32];  // [n][k] (transposed)

    const int tid  = threadIdx.x;
    const int w    = tid >> 5;
    const int lane = tid & 31;
    const int hgrp = lane >> 4;
    const int ln   = lane & 15;

    const int m0g = blockIdx.y * 128;
    const int n0g = blockIdx.x * 128;
    const int wm  = (w & 3) * 32;   // wave m offset in tile
    const int wn  = (w >> 2) * 64;  // wave n offset in tile

    v8f acc[2][4];
#pragma unroll
    for (int i = 0; i < 2; ++i)
#pragma unroll
        for (int j = 0; j < 4; ++j) acc[i][j] = {};

    for (int k0 = 0; k0 < K; k0 += 32) {
        // ---- stage A tile 128x32: each thread 32B via async b128 x2
        {
            int r = tid >> 1, c = (tid & 1) * 16;
            const unsigned short* gp = &A[(size_t)(m0g + r) * K + k0 + c];
            cp16_g2l(&As[r][c], gp);
            cp16_g2l(&As[r][c + 8], gp + 8);
        }
        // ---- stage B tile transposed: W[k][n] -> Bt[n][k]
        {
            int rk = tid >> 3, cn = (tid & 7) * 16;
            const unsigned short* gp = &W[(size_t)(k0 + rk) * N + n0g + cn];
            uint4 bv0 = *(const uint4*)gp;
            uint4 bv1 = *(const uint4*)(gp + 8);
            const unsigned short* bs0 = (const unsigned short*)&bv0;
            const unsigned short* bs1 = (const unsigned short*)&bv1;
#pragma unroll
            for (int j = 0; j < 8; ++j) Bt[cn + j][rk] = bs0[j];
#pragma unroll
            for (int j = 0; j < 8; ++j) Bt[cn + 8 + j][rk] = bs1[j];
        }
        cp_wait();
        __syncthreads();

        // ---- fragments (documented 16-bit layouts)
        VecBF a0, a1, b[4];
        {
            const unsigned short* rp = &As[wm + ln][0];
            a0.q[0] = *(const uint4*)(rp + hgrp * 8);
            a0.q[1] = *(const uint4*)(rp + 16 + hgrp * 8);
            rp = &As[wm + 16 + ln][0];
            a1.q[0] = *(const uint4*)(rp + hgrp * 8);
            a1.q[1] = *(const uint4*)(rp + 16 + hgrp * 8);
        }
#pragma unroll
        for (int j = 0; j < 4; ++j) {
            const unsigned short* rp = &Bt[wn + j * 16 + ln][0];
            b[j].q[0] = *(const uint4*)(rp + hgrp * 8);
            b[j].q[1] = *(const uint4*)(rp + 16 + hgrp * 8);
        }
#pragma unroll
        for (int j = 0; j < 4; ++j) {
            acc[0][j] = wmma_bf16(a0, b[j], acc[0][j]);
            acc[1][j] = wmma_bf16(a1, b[j], acc[1][j]);
        }
        __syncthreads();
    }

    // ---- epilogue: C layout VGPR v -> row m = v + 8*hgrp, col n = ln
#pragma unroll
    for (int mi = 0; mi < 2; ++mi) {
#pragma unroll
        for (int nj = 0; nj < 4; ++nj) {
            int n = n0g + wn + nj * 16 + ln;
            float bv = bias[n];
#pragma unroll
            for (int v = 0; v < 8; ++v) {
                int m = m0g + wm + mi * 16 + v + 8 * hgrp;
                float val = acc[mi][nj][v] + bv;
                if (RELU) val = fmaxf(val, 0.0f);
                size_t idx = (size_t)m * N + n;
                if (WB) outB[idx] = f2bf(val);
                if (WF) outF[idx] = val;
            }
        }
    }
}

// ---------------------------------------------------------------------------
// Flash attention (fused QK^T -> online softmax -> PV), WMMA bf16.
// Grid: (S/128, H, B). Block: 256 threads (8 waves); each wave owns 16 q rows.
// q/k/v/ctx layout: [b*S + s][h*64 + d]  (row stride 1024)
// ---------------------------------------------------------------------------
__global__ __launch_bounds__(256) void attn_kernel(const unsigned short* __restrict__ q,
                                                   const unsigned short* __restrict__ k,
                                                   const unsigned short* __restrict__ v,
                                                   unsigned short* __restrict__ ctx) {
    constexpr int S = 1024;
    constexpr int D = 1024;
    constexpr float INV_SCALE = 0.125f;  // 1/sqrt(64)

    __shared__ __align__(16) unsigned short Qs[128][64];  // [m][d]
    __shared__ __align__(16) unsigned short Ks[32][64];   // [key][d]
    __shared__ __align__(16) unsigned short Vt[64][32];   // [d][key] (transposed)
    __shared__ __align__(16) float pbuf[8][16][32];       // wave-private P staging

    const int tid  = threadIdx.x;
    const int w    = tid >> 5;
    const int lane = tid & 31;
    const int hgrp = lane >> 4;
    const int ln   = lane & 15;

    const int b  = blockIdx.z;
    const int hh = blockIdx.y;
    const int q0 = blockIdx.x * 128;
    const size_t base = (size_t)b * S * D + (size_t)hh * 64;

    // ---- stage Q tile (128x64): 32 bf16 per thread (async)
    {
        int r = tid >> 1, c = (tid & 1) * 32;
        const unsigned short* gp = &q[base + (size_t)(q0 + r) * D + c];
#pragma unroll
        for (int j = 0; j < 4; ++j) cp16_g2l(&Qs[r][c + j * 8], gp + j * 8);
    }

    VecBF qa0, qa1;
    float rmax[8], rsum[8];
#pragma unroll
    for (int i = 0; i < 8; ++i) { rmax[i] = -1e30f; rsum[i] = 0.0f; }
    v8f o0 = {}, o1 = {}, o2 = {}, o3 = {};

    cp_wait();
    __syncthreads();
    {
        const unsigned short* rp = &Qs[w * 16 + ln][0];
        qa0.q[0] = *(const uint4*)(rp + hgrp * 8);
        qa0.q[1] = *(const uint4*)(rp + 16 + hgrp * 8);
        qa1.q[0] = *(const uint4*)(rp + 32 + hgrp * 8);
        qa1.q[1] = *(const uint4*)(rp + 48 + hgrp * 8);
    }

    for (int kt = 0; kt < S; kt += 32) {
        __syncthreads();  // protect Ks/Vt from previous-iteration readers
        // ---- stage K tile [32 keys][64 d] (async) and V tile transposed [d][key]
        {
            int key = tid >> 3, c = (tid & 7) * 8;
            cp16_g2l(&Ks[key][c], &k[base + (size_t)(kt + key) * D + c]);
            uint4 vv = *(const uint4*)&v[base + (size_t)(kt + key) * D + c];
            const unsigned short* vs = (const unsigned short*)&vv;
#pragma unroll
            for (int j = 0; j < 8; ++j) Vt[c + j][key] = vs[j];
        }
        cp_wait();
        __syncthreads();

        // ---- scores = Q * K^T  (two 16x16 n-subtiles, two 32-wide d chunks)
        v8f s0 = {}, s1 = {};
#pragma unroll
        for (int dc = 0; dc < 2; ++dc) {
            const VecBF& a = dc ? qa1 : qa0;
            VecBF kb0, kb1;
            int off = dc * 32 + hgrp * 8;
            kb0.q[0] = *(const uint4*)&Ks[ln][off];
            kb0.q[1] = *(const uint4*)&Ks[ln][off + 16];
            kb1.q[0] = *(const uint4*)&Ks[16 + ln][off];
            kb1.q[1] = *(const uint4*)&Ks[16 + ln][off + 16];
            s0 = wmma_bf16(a, kb0, s0);
            s1 = wmma_bf16(a, kb1, s1);
        }

        // ---- online softmax per row (row m = vv + 8*hgrp lives in 16-lane group)
#pragma unroll
        for (int vv8 = 0; vv8 < 8; ++vv8) {
            float t0 = s0[vv8] * INV_SCALE;
            float t1 = s1[vv8] * INV_SCALE;
            float mx = red16_max(fmaxf(t0, t1));                // DPP, no LDS
            float nm = fmaxf(rmax[vv8], mx);
            float al = __expf(rmax[vv8] - nm);
            float p0 = __expf(t0 - nm);
            float p1 = __expf(t1 - nm);
            float ps = red16_sum(p0 + p1);                      // DPP, no LDS
            rsum[vv8] = rsum[vv8] * al + ps;
            rmax[vv8] = nm;
            o0[vv8] *= al; o1[vv8] *= al; o2[vv8] *= al; o3[vv8] *= al;
            int prow = vv8 + 8 * hgrp;
            pbuf[w][prow][ln]      = p0;
            pbuf[w][prow][16 + ln] = p1;
        }

        // ---- P (16x32) as bf16 A-fragment (wave-private LDS round trip)
        VecBF pa;
#pragma unroll
        for (int e = 0; e < 8; ++e)
            pa.s[e] = f2bf(pbuf[w][ln][hgrp * 8 + e]);
#pragma unroll
        for (int e = 8; e < 16; ++e)
            pa.s[e] = f2bf(pbuf[w][ln][16 + hgrp * 8 + (e - 8)]);

        // ---- O += P * V  (four 16-wide d-subtiles)
#pragma unroll
        for (int ns = 0; ns < 4; ++ns) {
            VecBF vb;
            const unsigned short* rp = &Vt[ns * 16 + ln][0];
            vb.q[0] = *(const uint4*)(rp + hgrp * 8);
            vb.q[1] = *(const uint4*)(rp + 16 + hgrp * 8);
            v8f* o = ns == 0 ? &o0 : ns == 1 ? &o1 : ns == 2 ? &o2 : &o3;
            *o = wmma_bf16(pa, vb, *o);
        }
    }

    // ---- normalize and write ctx (bf16)
#pragma unroll
    for (int vv8 = 0; vv8 < 8; ++vv8) {
        float inv = 1.0f / rsum[vv8];
        int m = q0 + w * 16 + vv8 + 8 * hgrp;
        size_t row = base + (size_t)m * D;
        ctx[row + 0 * 16 + ln] = f2bf(o0[vv8] * inv);
        ctx[row + 1 * 16 + ln] = f2bf(o1[vv8] * inv);
        ctx[row + 2 * 16 + ln] = f2bf(o2[vv8] * inv);
        ctx[row + 3 * 16 + ln] = f2bf(o3[vv8] * inv);
    }
}

// ---------------------------------------------------------------------------
// LayerNorm of (2*in) with gain/bias; write bf16 (+ optional f32).
// One block per row of 1024.
// ---------------------------------------------------------------------------
__global__ __launch_bounds__(256) void ln_kernel(const float* __restrict__ in,
                                                 const float* __restrict__ g,
                                                 const float* __restrict__ beta,
                                                 unsigned short* __restrict__ outB,
                                                 float* __restrict__ outF,
                                                 int writeF) {
    __shared__ float redA[8];
    __shared__ float redB[8];
    const int tid = threadIdx.x;
    const int w = tid >> 5, lane = tid & 31;
    const size_t row = (size_t)blockIdx.x * 1024;

    float t[4];
    float sa = 0.0f, sb = 0.0f;
#pragma unroll
    for (int j = 0; j < 4; ++j) {
        t[j] = 2.0f * in[row + tid * 4 + j];
        sa += t[j];
        sb += t[j] * t[j];
    }
#pragma unroll
    for (int off = 16; off; off >>= 1) {
        sa += __shfl_down(sa, off, 32);
        sb += __shfl_down(sb, off, 32);
    }
    if (lane == 0) { redA[w] = sa; redB[w] = sb; }
    __syncthreads();
    if (tid == 0) {
        float ta = 0.0f, tb = 0.0f;
#pragma unroll
        for (int i = 0; i < 8; ++i) { ta += redA[i]; tb += redB[i]; }
        redA[0] = ta; redB[0] = tb;
    }
    __syncthreads();
    float mu  = redA[0] * (1.0f / 1024.0f);
    float var = redB[0] * (1.0f / 1024.0f) - mu * mu;
    float rs  = rsqrtf(var + 1e-5f);
#pragma unroll
    for (int j = 0; j < 4; ++j) {
        int c = tid * 4 + j;
        float y = (t[j] - mu) * rs * g[c] + beta[c];
        outB[row + c] = f2bf(y);
        if (writeF) outF[row + c] = y;
    }
}

// ---------------------------------------------------------------------------
// Host launcher
// ---------------------------------------------------------------------------
extern "C" void kernel_launch(void* const* d_in, const int* in_sizes, int n_in,
                              void* d_out, int out_size, void* d_ws, size_t ws_size,
                              hipStream_t stream) {
    (void)in_sizes; (void)n_in; (void)out_size; (void)ws_size;
    constexpr int D = 1024, M = 4096, F = 4096;  // M = B*S

    const float* x   = (const float*)d_in[0];
    const float* Wq  = (const float*)d_in[1];  const float* bq  = (const float*)d_in[2];
    const float* Wk  = (const float*)d_in[3];  const float* bk  = (const float*)d_in[4];
    const float* Wv  = (const float*)d_in[5];  const float* bv  = (const float*)d_in[6];
    const float* Wo  = (const float*)d_in[7];  const float* bo  = (const float*)d_in[8];
    const float* W1  = (const float*)d_in[9];  const float* b1  = (const float*)d_in[10];
    const float* W2  = (const float*)d_in[11]; const float* b2  = (const float*)d_in[12];
    const float* g1  = (const float*)d_in[13]; const float* be1 = (const float*)d_in[14];
    const float* g2  = (const float*)d_in[15]; const float* be2 = (const float*)d_in[16];

    char* ws = (char*)d_ws;
    size_t off = 0;
    auto alloc = [&](size_t bytes) -> void* {
        void* p = ws + off;
        off += (bytes + 255) & ~(size_t)255;
        return p;
    };
    unsigned short* xb   = (unsigned short*)alloc((size_t)M * D * 2);
    unsigned short* qbuf = (unsigned short*)alloc((size_t)M * D * 2);
    unsigned short* kbuf = (unsigned short*)alloc((size_t)M * D * 2);
    unsigned short* vbuf = (unsigned short*)alloc((size_t)M * D * 2);
    unsigned short* ctx  = (unsigned short*)alloc((size_t)M * D * 2);
    unsigned short* hbuf = (unsigned short*)alloc((size_t)M * D * 2);
    unsigned short* ff1  = (unsigned short*)alloc((size_t)M * F * 2);
    float*          tmpF = (float*)alloc((size_t)M * D * 4);
    unsigned short* wqb  = (unsigned short*)alloc((size_t)D * D * 2);
    unsigned short* wkb  = (unsigned short*)alloc((size_t)D * D * 2);
    unsigned short* wvb  = (unsigned short*)alloc((size_t)D * D * 2);
    unsigned short* wob  = (unsigned short*)alloc((size_t)D * D * 2);
    unsigned short* w1b  = (unsigned short*)alloc((size_t)D * F * 2);
    unsigned short* w2b  = (unsigned short*)alloc((size_t)F * D * 2);

    auto cast = [&](const float* s, unsigned short* d, int n) {
        cast_bf16_kernel<<<(n + 255) / 256, 256, 0, stream>>>(s, d, n);
    };
    cast(x,  xb,  M * D);
    cast(Wq, wqb, D * D);
    cast(Wk, wkb, D * D);
    cast(Wv, wvb, D * D);
    cast(Wo, wob, D * D);
    cast(W1, w1b, D * F);
    cast(W2, w2b, F * D);

    dim3 blk(256);
    dim3 gD(D / 128, M / 128);   // N=1024 gemms: 8 x 32
    dim3 gF(F / 128, M / 128);   // N=4096 gemm: 32 x 32
    dim3 gAttn(1024 / 128, 16, 4);

    for (int L = 0; L < 12; ++L) {
        gemm_bf16_kernel<0, 0, 1><<<gD, blk, 0, stream>>>(xb, wqb, bq, qbuf, nullptr, M, D, D);
        gemm_bf16_kernel<0, 0, 1><<<gD, blk, 0, stream>>>(xb, wkb, bk, kbuf, nullptr, M, D, D);
        gemm_bf16_kernel<0, 0, 1><<<gD, blk, 0, stream>>>(xb, wvb, bv, vbuf, nullptr, M, D, D);

        attn_kernel<<<gAttn, blk, 0, stream>>>(qbuf, kbuf, vbuf, ctx);

        gemm_bf16_kernel<0, 1, 0><<<gD, blk, 0, stream>>>(ctx, wob, bo, nullptr, tmpF, M, D, D);
        ln_kernel<<<M, blk, 0, stream>>>(tmpF, g1, be1, hbuf, nullptr, 0);

        gemm_bf16_kernel<1, 0, 1><<<gF, blk, 0, stream>>>(hbuf, w1b, b1, ff1, nullptr, M, F, D);
        gemm_bf16_kernel<0, 1, 0><<<gD, blk, 0, stream>>>(ff1, w2b, b2, nullptr, tmpF, M, D, F);

        ln_kernel<<<M, blk, 0, stream>>>(tmpF, g2, be2, xb, (float*)d_out, (L == 11) ? 1 : 0);
    }
}
